// PointNet2SemSeg_33071248179388
// MI455X (gfx1250) — matrix operand order repaired and checked by reading, not compiled, not run on hardware
//
#include <hip/hip_runtime.h>
#include <hip/hip_bf16.h>
#include <math.h>

// ---------------------------------------------------------------------------
// PointNet++ semantic segmentation forward pass for gfx1250 (MI455X).
// Dense layers run through v_wmma_f32_16x16x32_f16. Activations on the GEMM
// path are kept in f16 (zero-padded to K%32==0), weights are pre-transposed
// into f16 (Nc x Kpad), so every fragment is a contiguous aligned vector load
// and the inner loop is 8 loads + 4 WMMA with no VALU and no branches.
// ---------------------------------------------------------------------------

typedef __attribute__((ext_vector_type(16))) _Float16 v16h;
typedef __attribute__((ext_vector_type(8)))  _Float16 v8h;
typedef __attribute__((ext_vector_type(8)))  float    v8f;

#define BN_EPS 1e-5f

static inline int cdiv(int a, int b) { return (a + b - 1) / b; }
static inline int pad32(int k) { return (k + 31) & ~31; }

// ---------------------------------------------------------------------------
// Farthest point sampling: one block per batch, running min-distance in LDS.
// ---------------------------------------------------------------------------
__global__ __launch_bounds__(256) void fps_kernel(const float* __restrict__ xyz,
                                                  int N, int npoint,
                                                  int* __restrict__ outIdx) {
  __shared__ float dist[8192];
  __shared__ float rval[256];
  __shared__ int   ridx[256];
  __shared__ int   sFar;
  const int b = blockIdx.x;
  const int t = threadIdx.x;
  const float* px = xyz + (size_t)b * N * 3;
  for (int n = t; n < N; n += 256) dist[n] = 1e10f;
  int far = 0;
  __syncthreads();
  for (int it = 0; it < npoint; ++it) {
    if (t == 0) outIdx[b * npoint + it] = far;
    const float cx = px[far * 3 + 0];
    const float cy = px[far * 3 + 1];
    const float cz = px[far * 3 + 2];
    float best = -1.0f; int bi = 0;
    for (int n = t; n < N; n += 256) {
      float dx = px[n * 3 + 0] - cx;
      float dy = px[n * 3 + 1] - cy;
      float dz = px[n * 3 + 2] - cz;
      float d  = dx * dx + dy * dy + dz * dz;
      float dn = fminf(dist[n], d);
      dist[n] = dn;
      if (dn > best) { best = dn; bi = n; }
    }
    rval[t] = best; ridx[t] = bi;
    __syncthreads();
    for (int s = 128; s > 0; s >>= 1) {
      if (t < s) {
        if (rval[t + s] > rval[t] ||
            (rval[t + s] == rval[t] && ridx[t + s] < ridx[t])) {
          rval[t] = rval[t + s]; ridx[t] = ridx[t + s];
        }
      }
      __syncthreads();
    }
    if (t == 0) sFar = ridx[0];
    __syncthreads();
    far = sFar;
  }
}

// Gather sampled centroid coordinates: new_xyz = xyz[fps_idx]
__global__ __launch_bounds__(256) void gather_xyz_kernel(const float* __restrict__ xyz,
                                                         const int* __restrict__ fpsIdx,
                                                         float* __restrict__ newxyz,
                                                         int B, int N, int S) {
  int t = blockIdx.x * blockDim.x + threadIdx.x;
  if (t >= B * S) return;
  int b = t / S;
  int idx = fpsIdx[t];
  const float* src = xyz + ((size_t)b * N + idx) * 3;
  float* dst = newxyz + (size_t)t * 3;
  dst[0] = src[0]; dst[1] = src[1]; dst[2] = src[2];
}

// Ball query: first `ns` indices (ascending) within radius; pad with first hit.
__global__ __launch_bounds__(128) void ball_query_kernel(const float* __restrict__ xyz,
                                                         const float* __restrict__ newxyz,
                                                         int* __restrict__ gidx,
                                                         int B, int N, int S, int ns,
                                                         float r2) {
  int t = blockIdx.x * blockDim.x + threadIdx.x;
  if (t >= B * S) return;
  int b = t / S;
  const float cx = newxyz[(size_t)t * 3 + 0];
  const float cy = newxyz[(size_t)t * 3 + 1];
  const float cz = newxyz[(size_t)t * 3 + 2];
  const float* px = xyz + (size_t)b * N * 3;
  int* g = gidx + (size_t)t * ns;
  int cnt = 0, first = -1;
  for (int n = 0; n < N; ++n) {
    float dx = px[n * 3 + 0] - cx;
    float dy = px[n * 3 + 1] - cy;
    float dz = px[n * 3 + 2] - cz;
    float d = dx * dx + dy * dy + dz * dz;
    if (d <= r2) {
      if (first < 0) first = n;
      g[cnt++] = n;
      if (cnt >= ns) break;
    }
  }
  if (first < 0) first = N - 1;
  for (int k = cnt; k < ns; ++k) g[k] = first;
}

// Grouped features (f16, padded row stride Kpad):
// row = [xyz[g]-center (3), points[g] (Cp), zeros (Kpad-3-Cp)]
__global__ __launch_bounds__(128) void group_kernel(const float* __restrict__ xyz,
                                                    const float* __restrict__ pts,
                                                    const int* __restrict__ gidx,
                                                    const float* __restrict__ newxyz,
                                                    _Float16* __restrict__ out,
                                                    int B, int N, int S, int ns,
                                                    int Cp, int Kpad) {
  int t = blockIdx.x * blockDim.x + threadIdx.x;
  if (t >= B * S * ns) return;
  int center = t / ns;        // b*S + s
  int b = center / S;
  int idx = gidx[t];
  const float* src = xyz + ((size_t)b * N + idx) * 3;
  const float* ctr = newxyz + (size_t)center * 3;
  _Float16* dst = out + (size_t)t * Kpad;
  dst[0] = (_Float16)(src[0] - ctr[0]);
  dst[1] = (_Float16)(src[1] - ctr[1]);
  dst[2] = (_Float16)(src[2] - ctr[2]);
  if (Cp > 0) {
    const float* psrc = pts + ((size_t)b * N + idx) * Cp;
    for (int c = 0; c < Cp; ++c) dst[3 + c] = (_Float16)psrc[c];
  }
  for (int c = 3 + Cp; c < Kpad; ++c) dst[c] = (_Float16)0.0f;
}

// Transpose + pack weights: W[K,Nc] (f32) -> Wt[Nc,Kpad] (f16, zero-padded).
__global__ __launch_bounds__(256) void prep_w_kernel(const float* __restrict__ W,
                                                     _Float16* __restrict__ Wt,
                                                     int K, int Kpad, int Nc) {
  int t = blockIdx.x * blockDim.x + threadIdx.x;
  if (t >= Nc * Kpad) return;
  int n = t / Kpad, kk = t % Kpad;
  Wt[t] = (kk < K) ? (_Float16)W[(size_t)kk * Nc + n] : (_Float16)0.0f;
}

// f32 -> f16 cast (for feeding a persistent f32 tensor into the GEMM path).
__global__ __launch_bounds__(256) void cast_f16_kernel(const float* __restrict__ X,
                                                       _Float16* __restrict__ Xh,
                                                       int total) {
  int t = blockIdx.x * blockDim.x + threadIdx.x;
  if (t >= total) return;
  Xh[t] = (_Float16)X[t];
}

// A fragment for one lane: two contiguous 8-half runs at +0 and +16 halves.
__device__ __forceinline__ v16h load_a_frag(const _Float16* __restrict__ p) {
  v8h lo = *(const v8h*)(p);
  v8h hi = *(const v8h*)(p + 16);
  return __builtin_shufflevector(lo, hi, 0, 1, 2, 3, 4, 5, 6, 7,
                                 8, 9, 10, 11, 12, 13, 14, 15);
}

// ---------------------------------------------------------------------------
// WMMA GEMM: Y[M,Nc] = A[M,lda(:K)] @ W[K,Nc] + bias, A and Wt in f16.
// Requirements (guaranteed by caller): M % 32 == 0, Nc % 32 == 0,
// Kpad % 32 == 0, lda >= Kpad with columns K..Kpad-1 of A zero-filled.
// One wave computes a 32x32 output tile: 2x2 fragments, 4 WMMA per k-step.
// ---------------------------------------------------------------------------
__global__ __launch_bounds__(256) void wmma_gemm_bias_kernel(const _Float16* __restrict__ A,
                                                             int lda,
                                                             const _Float16* __restrict__ Wt,
                                                             int Kpad,
                                                             const float* __restrict__ bias,
                                                             float* __restrict__ Y,
                                                             int M, int Nc) {
  const int lane = threadIdx.x & 31;
  const int wave = threadIdx.x >> 5;
  const int tN = Nc >> 5;
  const int tile = blockIdx.x * 8 + wave;
  if (tile >= (M >> 5) * tN) return;
  const int tm = tile / tN;
  const int tn = tile % tN;
  const int row = lane & 15;
  const int kg  = lane >> 4;

  const _Float16* a0p = A  + (size_t)(tm * 32 + row) * lda + kg * 8;
  const _Float16* a1p = a0p + (size_t)16 * lda;
  const _Float16* b0p = Wt + (size_t)(tn * 32 + row) * Kpad + kg * 8;
  const _Float16* b1p = b0p + (size_t)16 * Kpad;

  v8f c00 = {}, c01 = {}, c10 = {}, c11 = {};
  for (int k0 = 0; k0 < Kpad; k0 += 32) {
    v16h a0 = load_a_frag(a0p + k0);
    v16h a1 = load_a_frag(a1p + k0);
    v16h b0 = load_a_frag(b0p + k0);
    v16h b1 = load_a_frag(b1p + k0);
    c00 = __builtin_amdgcn_wmma_f32_16x16x32_f16(false, a0, false, b0, (short)0, c00, false, false);
    c01 = __builtin_amdgcn_wmma_f32_16x16x32_f16(false, a0, false, b1, (short)0, c01, false, false);
    c10 = __builtin_amdgcn_wmma_f32_16x16x32_f16(false, a1, false, b0, (short)0, c10, false, false);
    c11 = __builtin_amdgcn_wmma_f32_16x16x32_f16(false, a1, false, b1, (short)0, c11, false, false);
  }

  const int col = tn * 32 + row;
  const float bv0 = bias[col];
  const float bv1 = bias[col + 16];
  const int r0 = tm * 32 + kg * 8;
#pragma unroll
  for (int r = 0; r < 8; ++r) {
    Y[(size_t)(r0 + r)      * Nc + col]      = c00[r] + bv0;
    Y[(size_t)(r0 + r)      * Nc + col + 16] = c01[r] + bv1;
    Y[(size_t)(r0 + r + 16) * Nc + col]      = c10[r] + bv0;
    Y[(size_t)(r0 + r + 16) * Nc + col + 16] = c11[r] + bv1;
  }
}

// Batch-statistics per channel: mean & 1/sqrt(var+eps). One block per channel.
__global__ __launch_bounds__(256) void bn_stats_kernel(const float* __restrict__ Y,
                                                       int M, int Nc,
                                                       float* __restrict__ mean,
                                                       float* __restrict__ istd) {
  __shared__ float sh[256], sh2[256];
  const int c = blockIdx.x;
  const int t = threadIdx.x;
  float s = 0.0f, ss = 0.0f;
  for (int i = t; i < M; i += 256) {
    float v = Y[(size_t)i * Nc + c];
    s += v; ss += v * v;
  }
  sh[t] = s; sh2[t] = ss;
  __syncthreads();
  for (int k = 128; k > 0; k >>= 1) {
    if (t < k) { sh[t] += sh[t + k]; sh2[t] += sh2[t + k]; }
    __syncthreads();
  }
  if (t == 0) {
    float mu = sh[0] / (float)M;
    float var = sh2[0] / (float)M - mu * mu;
    mean[c] = mu;
    istd[c] = rsqrtf(var + BN_EPS);
  }
}

// Normalize + ReLU in place (f32) and emit an f16 mirror for the next GEMM.
__global__ __launch_bounds__(256) void bn_relu_kernel(float* __restrict__ Y,
                                                      _Float16* __restrict__ Yh,
                                                      const float* __restrict__ mean,
                                                      const float* __restrict__ istd,
                                                      const float* __restrict__ g,
                                                      const float* __restrict__ bt,
                                                      int total, int Nc) {
  int t = blockIdx.x * blockDim.x + threadIdx.x;
  if (t >= total) return;
  int c = t % Nc;
  float v = Y[t];
  v = fmaxf(g[c] * (v - mean[c]) * istd[c] + bt[c], 0.0f);
  Y[t]  = v;
  Yh[t] = (_Float16)v;
}

// Max over the nsample axis: feats[R, ns, C] -> out[R, C]
__global__ __launch_bounds__(256) void maxpool_kernel(const float* __restrict__ feats,
                                                      float* __restrict__ out,
                                                      int R, int ns, int C) {
  int t = blockIdx.x * blockDim.x + threadIdx.x;
  if (t >= R * C) return;
  int r = t / C, c = t % C;
  const float* src = feats + ((size_t)r * ns) * C + c;
  float m = -1e30f;
  for (int k = 0; k < ns; ++k) m = fmaxf(m, src[(size_t)k * C]);
  out[(size_t)r * C + c] = m;
}

// 3-NN search + inverse-distance weights.
__global__ __launch_bounds__(128) void knn3_kernel(const float* __restrict__ xyz1,
                                                   const float* __restrict__ xyz2,
                                                   int* __restrict__ idx3,
                                                   float* __restrict__ w3,
                                                   int B, int N1, int N2) {
  int t = blockIdx.x * blockDim.x + threadIdx.x;
  if (t >= B * N1) return;
  int b = t / N1;
  const float px = xyz1[(size_t)t * 3 + 0];
  const float py = xyz1[(size_t)t * 3 + 1];
  const float pz = xyz1[(size_t)t * 3 + 2];
  const float* q = xyz2 + (size_t)b * N2 * 3;
  float d0 = 1e30f, d1 = 1e30f, d2 = 1e30f;
  int   i0 = 0, i1 = 0, i2 = 0;
  for (int j = 0; j < N2; ++j) {
    float dx = q[j * 3 + 0] - px;
    float dy = q[j * 3 + 1] - py;
    float dz = q[j * 3 + 2] - pz;
    float d = dx * dx + dy * dy + dz * dz;
    if (d < d0)      { d2 = d1; i2 = i1; d1 = d0; i1 = i0; d0 = d; i0 = j; }
    else if (d < d1) { d2 = d1; i2 = i1; d1 = d; i1 = j; }
    else if (d < d2) { d2 = d; i2 = j; }
  }
  d0 = fmaxf(d0, 1e-10f); d1 = fmaxf(d1, 1e-10f); d2 = fmaxf(d2, 1e-10f);
  float w0 = 1.0f / d0, w1 = 1.0f / d1, w2 = 1.0f / d2;
  float inv = 1.0f / (w0 + w1 + w2);
  idx3[t * 3 + 0] = i0; idx3[t * 3 + 1] = i1; idx3[t * 3 + 2] = i2;
  w3[t * 3 + 0] = w0 * inv; w3[t * 3 + 1] = w1 * inv; w3[t * 3 + 2] = w2 * inv;
}

// out[row] = concat(points1[row], sum_j w_j * points2[b, idx_j])  (f16 output;
// C1 + C2 is a multiple of 32 for every FP stage, so no padding is needed.)
__global__ __launch_bounds__(128) void interp_concat_kernel(const float* __restrict__ pts1, int C1,
                                                            const float* __restrict__ pts2, int C2,
                                                            const int* __restrict__ idx3,
                                                            const float* __restrict__ w3,
                                                            _Float16* __restrict__ out,
                                                            int B, int N1, int N2) {
  int t = blockIdx.x * blockDim.x + threadIdx.x;
  if (t >= B * N1) return;
  int b = t / N1;
  _Float16* dst = out + (size_t)t * (C1 + C2);
  if (C1 > 0) {
    const float* s1 = pts1 + (size_t)t * C1;
    for (int c = 0; c < C1; ++c) dst[c] = (_Float16)s1[c];
  }
  int i0 = idx3[t * 3 + 0], i1 = idx3[t * 3 + 1], i2 = idx3[t * 3 + 2];
  float w0 = w3[t * 3 + 0], w1 = w3[t * 3 + 1], w2 = w3[t * 3 + 2];
  const float* p0 = pts2 + ((size_t)b * N2 + i0) * C2;
  const float* p1 = pts2 + ((size_t)b * N2 + i1) * C2;
  const float* p2 = pts2 + ((size_t)b * N2 + i2) * C2;
  for (int c = 0; c < C2; ++c)
    dst[C1 + c] = (_Float16)(w0 * p0[c] + w1 * p1[c] + w2 * p2[c]);
}

// Final 128->13 linear + (B, N, 13) -> (B, 13, N) transpose.
__global__ __launch_bounds__(128) void head2_kernel(const float* __restrict__ X,
                                                    const float* __restrict__ W2,
                                                    const float* __restrict__ b2,
                                                    float* __restrict__ out,
                                                    int B, int N) {
  int t = blockIdx.x * blockDim.x + threadIdx.x;
  if (t >= B * N) return;
  int b = t / N, n = t % N;
  const float* x = X + (size_t)t * 128;
  for (int c = 0; c < 13; ++c) {
    float acc = b2[c];
    for (int k = 0; k < 128; ++k) acc += x[k] * W2[k * 13 + c];
    out[((size_t)b * 13 + c) * N + n] = acc;
  }
}

// ---------------------------------------------------------------------------
// Host-side orchestration
// ---------------------------------------------------------------------------
struct Layer { const float *W, *b, *g, *bt; int cin, cout; };

// inH: f16 activations (stride lda0). Returns f32 pointer of final layer.
static const float* run_mlp(const Layer* L, int nL, const _Float16* inH, int lda0,
                            float* s0, float* s1, _Float16* h0, _Float16* h1,
                            _Float16* wt, int M, float* bnM, float* bnI,
                            hipStream_t st) {
  const _Float16* curH = inH;
  const float* curF = nullptr;
  int lda = lda0;
  int sel = (inH == h0) ? 1 : 0;
  for (int i = 0; i < nL; ++i) {
    const int K = L[i].cin, Kpad = pad32(K), Nc = L[i].cout;
    float*    dst  = sel ? s1 : s0;
    _Float16* dstH = sel ? h1 : h0;
    prep_w_kernel<<<cdiv(Nc * Kpad, 256), 256, 0, st>>>(L[i].W, wt, K, Kpad, Nc);
    int tiles = (M >> 5) * (Nc >> 5);
    wmma_gemm_bias_kernel<<<cdiv(tiles, 8), 256, 0, st>>>(curH, lda, wt, Kpad,
                                                          L[i].b, dst, M, Nc);
    bn_stats_kernel<<<Nc, 256, 0, st>>>(dst, M, Nc, bnM, bnI);
    int tot = M * Nc;
    bn_relu_kernel<<<cdiv(tot, 256), 256, 0, st>>>(dst, dstH, bnM, bnI,
                                                   L[i].g, L[i].bt, tot, Nc);
    curH = dstH; curF = dst;
    lda = Nc;
    sel ^= 1;
  }
  return curF;
}

static void set_abstraction_stage(const float* xyz, const float* pts, int Cp,
                                  int B, int N, int S, int ns, float radius,
                                  const Layer* L,
                                  float* newxyz, float* pooled,
                                  float* ping, float* pong,
                                  _Float16* hping, _Float16* hpong, _Float16* wt,
                                  int* fpsIdx, int* gidx,
                                  float* bnM, float* bnI, hipStream_t st) {
  const int Kpad = pad32(3 + Cp);
  fps_kernel<<<B, 256, 0, st>>>(xyz, N, S, fpsIdx);
  gather_xyz_kernel<<<cdiv(B * S, 256), 256, 0, st>>>(xyz, fpsIdx, newxyz, B, N, S);
  ball_query_kernel<<<cdiv(B * S, 128), 128, 0, st>>>(xyz, newxyz, gidx, B, N, S, ns,
                                                      radius * radius);
  group_kernel<<<cdiv(B * S * ns, 128), 128, 0, st>>>(xyz, pts, gidx, newxyz, hping,
                                                      B, N, S, ns, Cp, Kpad);
  int M = B * S * ns;
  const float* f = run_mlp(L, 3, hping, Kpad, ping, pong, hping, hpong, wt, M,
                           bnM, bnI, st);
  int C = L[2].cout;
  maxpool_kernel<<<cdiv(B * S * C, 256), 256, 0, st>>>(f, pooled, B * S, ns, C);
}

static void feature_prop_stage(const float* xyz1, int N1, const float* xyz2, int N2,
                               const float* pts1, int C1, const float* pts2, int C2,
                               const Layer* L, int nL, float* outbuf,
                               float* ping, float* pong,
                               _Float16* hping, _Float16* hpong, _Float16* wt,
                               int* knnIdx, float* knnW,
                               float* bnM, float* bnI, hipStream_t st) {
  const int B = 8;
  knn3_kernel<<<cdiv(B * N1, 128), 128, 0, st>>>(xyz1, xyz2, knnIdx, knnW, B, N1, N2);
  interp_concat_kernel<<<cdiv(B * N1, 128), 128, 0, st>>>(pts1, C1, pts2, C2,
                                                          knnIdx, knnW, hping, B, N1, N2);
  const float* f = run_mlp(L, nL, hping, C1 + C2, ping, pong, hping, hpong, wt,
                           B * N1, bnM, bnI, st);
  (void)hipMemcpyAsync(outbuf, f, (size_t)B * N1 * L[nL - 1].cout * sizeof(float),
                       hipMemcpyDeviceToDevice, st);
}

extern "C" void kernel_launch(void* const* d_in, const int* in_sizes, int n_in,
                              void* d_out, int out_size, void* d_ws, size_t ws_size,
                              hipStream_t stream) {
  (void)in_sizes; (void)n_in; (void)out_size; (void)ws_size;
  const int B = 8, N0 = 8192;

  // ---- unpack inputs: pytree sorted-key order: params{fp1..fp4,head1,head2,sa1..sa4}, xyz
  int p = 0;
  auto nextp = [&]() -> const float* { return (const float*)d_in[p++]; };
  Layer fp1[3], fp2[2], fp3[2], fp4[2], head1[1], sa1[3], sa2[3], sa3[3], sa4[3];
  auto readMlp = [&](Layer* L, int n, int cin, const int* outs) {
    for (int i = 0; i < n; ++i) {
      L[i].W = nextp(); L[i].b = nextp(); L[i].g = nextp(); L[i].bt = nextp();
      L[i].cin = cin; L[i].cout = outs[i]; cin = outs[i];
    }
  };
  { int o[3] = {128, 128, 128}; readMlp(fp1, 3, 128, o); }
  { int o[2] = {256, 128};      readMlp(fp2, 2, 320, o); }
  { int o[2] = {256, 256};      readMlp(fp3, 2, 384, o); }
  { int o[2] = {256, 256};      readMlp(fp4, 2, 768, o); }
  { int o[1] = {128};           readMlp(head1, 1, 128, o); }
  const float* W2  = nextp();
  const float* b2v = nextp();
  { int o[3] = {32, 32, 64};    readMlp(sa1, 3, 3,   o); }
  { int o[3] = {64, 64, 128};   readMlp(sa2, 3, 67,  o); }
  { int o[3] = {128, 128, 256}; readMlp(sa3, 3, 131, o); }
  { int o[3] = {256, 256, 512}; readMlp(sa4, 3, 259, o); }
  const float* xyz = nextp();

  // ---- workspace layout
  char* base = (char*)d_ws;
  size_t off = 0;
  auto alloc = [&](size_t bytes) -> void* {
    off = (off + 255) & ~(size_t)255;
    void* q = base + off; off += bytes; return q;
  };
  float*     ping   = (float*)alloc((size_t)16777216 * 4);      // 262144 x 64 max (f32)
  float*     pong   = (float*)alloc((size_t)16777216 * 4);
  _Float16*  hping  = (_Float16*)alloc((size_t)16777216 * 2);   // f16 mirrors
  _Float16*  hpong  = (_Float16*)alloc((size_t)16777216 * 2);
  _Float16*  wt     = (_Float16*)alloc((size_t)262144 * 2);     // max Nc*Kpad = 256*768
  int*       fpsIdx = (int*)alloc((size_t)8192 * 4);
  int*       gidx   = (int*)alloc((size_t)262144 * 4);
  float* l1x = (float*)alloc((size_t)B * 1024 * 3 * 4);
  float* l2x = (float*)alloc((size_t)B * 256  * 3 * 4);
  float* l3x = (float*)alloc((size_t)B * 64   * 3 * 4);
  float* l4x = (float*)alloc((size_t)B * 16   * 3 * 4);
  float* l1p = (float*)alloc((size_t)B * 1024 * 64  * 4);
  float* l2p = (float*)alloc((size_t)B * 256  * 128 * 4);
  float* l3p = (float*)alloc((size_t)B * 64   * 256 * 4);
  float* l4p = (float*)alloc((size_t)B * 16   * 512 * 4);
  float* fp4o = (float*)alloc((size_t)B * 64   * 256 * 4);
  float* fp3o = (float*)alloc((size_t)B * 256  * 256 * 4);
  float* fp2o = (float*)alloc((size_t)B * 1024 * 128 * 4);
  float* fp1o = (float*)alloc((size_t)B * 8192 * 128 * 4);
  int*   knnIdx = (int*)  alloc((size_t)B * 8192 * 3 * 4);
  float* knnW   = (float*)alloc((size_t)B * 8192 * 3 * 4);
  float* bnM = (float*)alloc(512 * 4);
  float* bnI = (float*)alloc(512 * 4);

  // ---- encoder (set abstraction)
  set_abstraction_stage(xyz, nullptr, 0,  B, N0,   1024, 32, 0.1f, sa1,
                        l1x, l1p, ping, pong, hping, hpong, wt,
                        fpsIdx, gidx, bnM, bnI, stream);
  set_abstraction_stage(l1x, l1p, 64,     B, 1024, 256,  32, 0.2f, sa2,
                        l2x, l2p, ping, pong, hping, hpong, wt,
                        fpsIdx, gidx, bnM, bnI, stream);
  set_abstraction_stage(l2x, l2p, 128,    B, 256,  64,   32, 0.4f, sa3,
                        l3x, l3p, ping, pong, hping, hpong, wt,
                        fpsIdx, gidx, bnM, bnI, stream);
  set_abstraction_stage(l3x, l3p, 256,    B, 64,   16,   32, 0.8f, sa4,
                        l4x, l4p, ping, pong, hping, hpong, wt,
                        fpsIdx, gidx, bnM, bnI, stream);

  // ---- decoder (feature propagation)
  feature_prop_stage(l3x, 64,   l4x, 16,   l3p, 256, l4p, 512, fp4, 2, fp4o,
                     ping, pong, hping, hpong, wt, knnIdx, knnW, bnM, bnI, stream);
  feature_prop_stage(l2x, 256,  l3x, 64,   l2p, 128, fp4o, 256, fp3, 2, fp3o,
                     ping, pong, hping, hpong, wt, knnIdx, knnW, bnM, bnI, stream);
  feature_prop_stage(l1x, 1024, l2x, 256,  l1p, 64,  fp3o, 256, fp2, 2, fp2o,
                     ping, pong, hping, hpong, wt, knnIdx, knnW, bnM, bnI, stream);
  feature_prop_stage(xyz, 8192, l1x, 1024, nullptr, 0, fp2o, 128, fp1, 3, fp1o,
                     ping, pong, hping, hpong, wt, knnIdx, knnW, bnM, bnI, stream);

  // ---- head
  cast_f16_kernel<<<cdiv(B * 8192 * 128, 256), 256, 0, stream>>>(fp1o, hping,
                                                                 B * 8192 * 128);
  const float* h = run_mlp(head1, 1, hping, 128, ping, pong, hping, hpong, wt,
                           B * 8192, bnM, bnI, stream);
  head2_kernel<<<cdiv(B * 8192, 128), 128, 0, stream>>>(h, W2, b2v, (float*)d_out,
                                                        B, 8192);
}
